// SGConv2D_50620484550871
// MI455X (gfx1250) — matrix operand order repaired and checked by reading, not compiled, not run on hardware
//
#include <hip/hip_runtime.h>
#include <hip/hip_bf16.h>

// ---------------------------------------------------------------------------
// Problem constants (fixed by the reference): B=8, C=256, H=W=48, N=2304
// ---------------------------------------------------------------------------
#define BATCH   8
#define CH      256
#define HH      48
#define WW      48
#define NN      (HH * WW)        // 2304
#define ITILES  (NN / 16)        // 144 row tiles of 16
#define KSTEPS  (CH / 32)        // 8 k-steps of 32 over the channel dim
#define MBLKS   (NN / 32)        // 72 k-steps of 32 over the node dim

#define THRESH  0.01f

typedef __attribute__((ext_vector_type(16))) __bf16 v16bf;
typedef __attribute__((ext_vector_type(8)))  __bf16 v8bf;
typedef __attribute__((ext_vector_type(8)))  float  v8f;

// LDS strides padded to avoid bank conflicts on row-indexed reads.
// LSTRH in fp16 elements: LSTRH/2 = 1153 == 1 (mod 64)  -> distinct banks/row
// NSTR  in fp32 elements: NSTR  == 4 (mod 64)           -> distinct banks/row
#define LSTRH 2306               // logits row stride (fp16 elements)
#define NSTR  260                // nbr row stride (fp32 elements)

// ---------------------------------------------------------------------------
// WMMA wrapper: D = A(16x32 bf16) * B(32x16 bf16) + C(16x16 f32)
// ---------------------------------------------------------------------------
__device__ __forceinline__ v8f wmma_bf16(v16bf a, v16bf b, v8f c) {
    return __builtin_amdgcn_wmma_f32_16x16x32_bf16(
        /*neg_a=*/false, a, /*neg_b=*/false, b,
        /*c_mod=*/(short)0, c, /*reuse_a=*/false, /*reuse_b=*/false);
}

// A-tile (16x32, M x K) from a row-major bf16 matrix.
// ISA layout: lanes 0-15 hold row (lane), K elems 0..7 & 16..23;
//             lanes 16-31 hold row (lane-16), K elems 8..15 & 24..31.
__device__ __forceinline__ v16bf load_a_tile(const __bf16* base, int row_stride,
                                             int r0, int k0) {
    const int lane  = threadIdx.x & 31;
    const int row   = r0 + (lane & 15);
    const int cbase = k0 + ((lane & 16) ? 8 : 0);
    const __bf16* p = base + (long)row * row_stride + cbase;
    v8bf lo = *(const v8bf*)(p);
    v8bf hi = *(const v8bf*)(p + 16);
    v16bf a;
#pragma unroll
    for (int e = 0; e < 8; ++e) { a[e] = lo[e]; a[8 + e] = hi[e]; }
    return a;
}

// A-tile built from an fp16 LDS array (converted to bf16).
__device__ __forceinline__ v16bf a_tile_from_lds_h(const _Float16* lds, int row_stride,
                                                   int k0) {
    const int lane  = threadIdx.x & 31;
    const int row   = lane & 15;
    const int cbase = k0 + ((lane & 16) ? 8 : 0);
    const _Float16* p = lds + row * row_stride + cbase;
    v16bf a;
#pragma unroll
    for (int e = 0; e < 8; ++e) {
        a[e]     = (__bf16)(float)p[e];
        a[8 + e] = (__bf16)(float)p[16 + e];
    }
    return a;
}

// A-tile built from an fp32 LDS array.
__device__ __forceinline__ v16bf a_tile_from_lds_f(const float* lds, int row_stride,
                                                   int k0) {
    const int lane  = threadIdx.x & 31;
    const int row   = lane & 15;
    const int cbase = k0 + ((lane & 16) ? 8 : 0);
    const float* p  = lds + row * row_stride + cbase;
    v16bf a;
#pragma unroll
    for (int e = 0; e < 8; ++e) {
        a[e]     = (__bf16)p[e];
        a[8 + e] = (__bf16)p[16 + e];
    }
    return a;
}

// A-tile from row-major bf16 global memory scaled by per-row diag value.
__device__ __forceinline__ v16bf load_a_tile_scaled(const __bf16* base, int row_stride,
                                                    int r0, int k0, const float* diag) {
    const int lane  = threadIdx.x & 31;
    const int rloc  = lane & 15;
    const int row   = r0 + rloc;
    const int cbase = k0 + ((lane & 16) ? 8 : 0);
    const float d   = diag[rloc];
    const __bf16* p = base + (long)row * row_stride + cbase;
    v8bf lo = *(const v8bf*)(p);
    v8bf hi = *(const v8bf*)(p + 16);
    v16bf a;
#pragma unroll
    for (int e = 0; e < 8; ++e) {
        a[e]     = (__bf16)((float)lo[e] * d);
        a[8 + e] = (__bf16)((float)hi[e] * d);
    }
    return a;
}

// B-tile (32x16, K x N) where column n of B is row (n0+n) of a row-major bf16
// matrix, K running along that row from k0.
// ISA layout: lane n (0-15) holds column n, K 0..7 & 16..23; lane n+16 holds column n, K 8..15 & 24..31.
__device__ __forceinline__ v16bf load_b_tile(const __bf16* base, int row_stride,
                                             int n0, int k0) {
    const int lane  = threadIdx.x & 31;
    const int row   = n0 + (lane & 15);
    const int kbase = k0 + ((lane & 16) ? 8 : 0);
    const __bf16* p = base + (long)row * row_stride + kbase;
    v8bf lo = *(const v8bf*)(p);
    v8bf hi = *(const v8bf*)(p + 16);
    v16bf bv;
#pragma unroll
    for (int e = 0; e < 8; ++e) { bv[e] = lo[e]; bv[8 + e] = hi[e]; }
    return bv;
}

// ---------------------------------------------------------------------------
// Kernel 1a: x (B,C,N) fp32 -> XbfT (B,C,N) bf16 (copy) and Xbf (B,N,C) bf16.
// LDS 32x32 tile transpose so both stores are contiguous.
// ---------------------------------------------------------------------------
__global__ __launch_bounds__(256)
void pack_x_kernel(const float* __restrict__ x,
                   __bf16* __restrict__ XbfT,
                   __bf16* __restrict__ Xbf) {
    __shared__ __bf16 tile[32][33];
    const int tilesPerB = (CH / 32) * (NN / 32);
    const int bid = blockIdx.x;
    const int b   = bid / tilesPerB;
    const int rem = bid % tilesPerB;
    const int c0  = (rem / (NN / 32)) * 32;
    const int n0  = (rem % (NN / 32)) * 32;

    const int j = threadIdx.x & 31;       // fast index within a row
    const int i = threadIdx.x >> 5;       // 0..7

    const float* xb = x + (long)b * CH * NN;
    __bf16* xtb = XbfT + (long)b * CH * NN;
#pragma unroll
    for (int ii = i; ii < 32; ii += 8) {
        const __bf16 bv = (__bf16)xb[(long)(c0 + ii) * NN + n0 + j];
        xtb[(long)(c0 + ii) * NN + n0 + j] = bv;    // contiguous along n
        tile[ii][j] = bv;
    }
    __syncthreads();
    __bf16* xnb = Xbf + (long)b * NN * CH;
#pragma unroll
    for (int ii = i; ii < 32; ii += 8) {
        xnb[(long)(n0 + ii) * CH + c0 + j] = tile[j][ii];   // contiguous along c
    }
}

// ---------------------------------------------------------------------------
// Kernel 1b: convert the 4 weight matrices (C x C, fp32, row-major) to bf16.
// ---------------------------------------------------------------------------
__global__ void pack_w_kernel(const float* __restrict__ phi,
                              const float* __restrict__ psi,
                              const float* __restrict__ w0,
                              const float* __restrict__ w1,
                              __bf16* __restrict__ dst) {
    const int total = 4 * CH * CH;
    int idx = blockIdx.x * blockDim.x + threadIdx.x;
    if (idx >= total) return;
    const int m = idx >> 16;           // CH*CH == 65536
    const int r = idx & 65535;
    const float* src = (m == 0) ? phi : (m == 1) ? psi : (m == 2) ? w0 : w1;
    dst[idx] = (__bf16)src[r];
}

// ---------------------------------------------------------------------------
// Kernel 2: Q = X @ phi^T, K = X @ psi^T  (bf16 in, bf16 out, f32 accum)
// Each wave advances its 4 d-tiles together (4 accumulator chains) with the
// next k-step's 4 B-tiles prefetched before the current 4 WMMAs.
// ---------------------------------------------------------------------------
__global__ __launch_bounds__(128)
void qk_gemm_kernel(const __bf16* __restrict__ Xbf,
                    const __bf16* __restrict__ phibf,
                    const __bf16* __restrict__ psibf,
                    __bf16* __restrict__ Qbf,
                    __bf16* __restrict__ Kbf) {
    const int bid  = blockIdx.x;
    const int b    = bid / ITILES;
    const int i0   = (bid % ITILES) * 16;
    const int wv   = threadIdx.x >> 5;
    const int lane = threadIdx.x & 31;

    const __bf16* Xb = Xbf + (long)b * NN * CH;

    // Preload the 8 A-tiles of X for this row tile (shared by Q and K).
    v16bf xa[KSTEPS];
#pragma unroll
    for (int k = 0; k < KSTEPS; ++k) xa[k] = load_a_tile(Xb, CH, i0, k * 32);

#pragma unroll
    for (int which = 0; which < 2; ++which) {
        const __bf16* wmat = (which == 0) ? phibf : psibf;
        __bf16* dstbase    = ((which == 0) ? Qbf : Kbf) + (long)b * NN * CH;

        v8f acc[4];
#pragma unroll
        for (int q = 0; q < 4; ++q) acc[q] = (v8f){};

        v16bf bt[4];
#pragma unroll
        for (int q = 0; q < 4; ++q)
            bt[q] = load_b_tile(wmat, CH, (wv * 4 + q) * 16, 0);

#pragma unroll
        for (int k = 0; k < KSTEPS; ++k) {
            v16bf nb[4];
            if (k + 1 < KSTEPS) {                     // prefetch next k-step
#pragma unroll
                for (int q = 0; q < 4; ++q)
                    nb[q] = load_b_tile(wmat, CH, (wv * 4 + q) * 16, (k + 1) * 32);
            }
#pragma unroll
            for (int q = 0; q < 4; ++q) acc[q] = wmma_bf16(xa[k], bt[q], acc[q]);
            if (k + 1 < KSTEPS) {
#pragma unroll
                for (int q = 0; q < 4; ++q) bt[q] = nb[q];
            }
        }
        // D layout: element g -> row g (lanes 0-15) / g+8 (lanes 16-31)
#pragma unroll
        for (int q = 0; q < 4; ++q) {
            const int d0 = (wv * 4 + q) * 16;
#pragma unroll
            for (int g = 0; g < 8; ++g) {
                const int row = i0 + g + ((lane & 16) ? 8 : 0);
                dstbase[(long)row * CH + d0 + (lane & 15)] = (__bf16)acc[q][g];
            }
        }
    }
}

// ---------------------------------------------------------------------------
// Kernel 3: fused attention-graph kernel.
// One block per (batch, 16-row tile); 128 threads (4 waves).
// Logits scratch is fp16 (74 KB) -> 4 blocks per 320 KB WGP -> 4 waves/SIMD,
// so other waves issue WMMAs while one wave sits on a load wait.
//   Phase 1: logits(16 x N) = Q_tile @ K^T  -> fp16 LDS  (4 chains + prefetch)
//   Phase 2: row softmax, threshold, +grid adjacency, extract diag, zero diag
//   Phase 3: nbr(16 x C) = A_off @ X                     (4 chains + prefetch)
//   Phase 4: Y = relu( (X*diag) @ w0^T + nbr @ w1^T ), staged & coalesced out
// ---------------------------------------------------------------------------
__global__ __launch_bounds__(128)
void fused_graph_kernel(const __bf16* __restrict__ Qbf,
                        const __bf16* __restrict__ Kbf,
                        const __bf16* __restrict__ Xbf,
                        const __bf16* __restrict__ XbfT,
                        const __bf16* __restrict__ w0bf,
                        const __bf16* __restrict__ w1bf,
                        float* __restrict__ out) {
    // 16 x LSTRH fp16 logits; after phase 3 the same bytes hold fp32 nbr + y.
    __shared__ __attribute__((aligned(16))) char smem[16 * LSTRH * 2];  // 73,792 B
    __shared__ float lds_red[16 * 8];
    __shared__ float lds_diag[16];

    _Float16* lds_log = (_Float16*)smem;            // 16 x LSTRH (fp16)
    float*    lds_nbr = (float*)smem;               // 16 x NSTR  (fp32, overlay)
    float*    lds_y   = (float*)smem + 16 * NSTR;   // 256 x 16   (fp32, overlay)

    const int bid  = blockIdx.x;
    const int b    = bid / ITILES;
    const int i0   = (bid % ITILES) * 16;
    const int wv   = threadIdx.x >> 5;
    const int lane = threadIdx.x & 31;

    const __bf16* Qb  = Qbf  + (long)b * NN * CH;
    const __bf16* Kb  = Kbf  + (long)b * NN * CH;
    const __bf16* Xb  = Xbf  + (long)b * NN * CH;
    const __bf16* XTb = XbfT + (long)b * CH * NN;

    // ---------------- Phase 1: logits = Q_tile @ K^T ----------------
    v16bf qa[KSTEPS];
#pragma unroll
    for (int k = 0; k < KSTEPS; ++k) qa[k] = load_a_tile(Qb, CH, i0, k * 32);

    // Each wave advances 4 column tiles (tb+wv, +4, +8, +12) together:
    // 4 independent accumulator chains, next-k B-tiles prefetched.
    for (int tb = 0; tb < ITILES; tb += 16) {         // 9 iterations
        v8f acc[4];
#pragma unroll
        for (int q = 0; q < 4; ++q) acc[q] = (v8f){};

        v16bf bt[4];
#pragma unroll
        for (int q = 0; q < 4; ++q)
            bt[q] = load_b_tile(Kb, CH, (tb + wv + 4 * q) * 16, 0);

#pragma unroll
        for (int k = 0; k < KSTEPS; ++k) {
            v16bf nb[4];
            if (k + 1 < KSTEPS) {
#pragma unroll
                for (int q = 0; q < 4; ++q)
                    nb[q] = load_b_tile(Kb, CH, (tb + wv + 4 * q) * 16, (k + 1) * 32);
            }
#pragma unroll
            for (int q = 0; q < 4; ++q) acc[q] = wmma_bf16(qa[k], bt[q], acc[q]);
            if (k + 1 < KSTEPS) {
#pragma unroll
                for (int q = 0; q < 4; ++q) bt[q] = nb[q];
            }
        }
#pragma unroll
        for (int q = 0; q < 4; ++q) {
            const int n0 = (tb + wv + 4 * q) * 16;
#pragma unroll
            for (int g = 0; g < 8; ++g) {
                const int r = g + ((lane & 16) ? 8 : 0);
                lds_log[r * LSTRH + n0 + (lane & 15)] = (_Float16)acc[q][g];
            }
        }
    }
    __syncthreads();

    // ---------------- Phase 2: softmax + threshold + adjacency ----------------
    {
        const int row = threadIdx.x >> 3;   // 0..15
        const int sub = threadIdx.x & 7;    // 8 threads per row
        _Float16* rp = lds_log + row * LSTRH;

        float m = -3.0e38f;
        for (int j = sub; j < NN; j += 8) m = fmaxf(m, (float)rp[j]);
        lds_red[row * 8 + sub] = m;
        __syncthreads();
        float rmax = lds_red[row * 8];
#pragma unroll
        for (int t = 1; t < 8; ++t) rmax = fmaxf(rmax, lds_red[row * 8 + t]);
        __syncthreads();

        float s = 0.0f;
        for (int j = sub; j < NN; j += 8) {
            const float e = __expf((float)rp[j] - rmax);
            rp[j] = (_Float16)e;
            s += e;
        }
        lds_red[row * 8 + sub] = s;
        __syncthreads();
        float rsum = 0.0f;
#pragma unroll
        for (int t = 0; t < 8; ++t) rsum += lds_red[row * 8 + t];
        const float inv = 1.0f / rsum;

        const int node = i0 + row;
        const int h = node / WW;
        const int w = node - h * WW;
        for (int j = sub; j < NN; j += 8) {
            float v = (float)rp[j] * inv;
            if (v < THRESH) v = 0.0f;             // threshold S first
            if (j == node) {
                lds_diag[row] = 1.0f + v;         // A_base diag (self loop) + S_ii
                v = 0.0f;                         // A_off zeroes the diagonal
            } else {
                const bool adj = (j == node - WW && h > 0) ||
                                 (j == node + WW && h < HH - 1) ||
                                 (j == node - 1  && w > 0) ||
                                 (j == node + 1  && w < WW - 1);
                if (adj) v += 1.0f;               // + A_base off-diagonal
            }
            rp[j] = (_Float16)v;
        }
    }
    __syncthreads();

    // ---------------- Phase 3: nbr = A_off @ X ----------------
    v8f acc4[4];
#pragma unroll
    for (int q = 0; q < 4; ++q) acc4[q] = (v8f){};

    v16bf bt[4];
#pragma unroll
    for (int q = 0; q < 4; ++q)
        bt[q] = load_b_tile(XTb, NN, (wv * 4 + q) * 16, 0);

    for (int mb = 0; mb < MBLKS; ++mb) {
        const int m0 = mb * 32;
        v16bf at = a_tile_from_lds_h(lds_log, LSTRH, m0);  // A_off (fp16->bf16)
        v16bf nb[4];
        const bool more = (mb + 1 < MBLKS);
        if (more) {                                       // prefetch next m-block
#pragma unroll
            for (int q = 0; q < 4; ++q)
                nb[q] = load_b_tile(XTb, NN, (wv * 4 + q) * 16, m0 + 32);
        }
#pragma unroll
        for (int q = 0; q < 4; ++q) acc4[q] = wmma_bf16(at, bt[q], acc4[q]);
        if (more) {
#pragma unroll
            for (int q = 0; q < 4; ++q) bt[q] = nb[q];
        }
    }
    __syncthreads();   // A_off reads done -> safe to overlay nbr onto smem

#pragma unroll
    for (int q = 0; q < 4; ++q) {
        const int c0 = (wv * 4 + q) * 16;
#pragma unroll
        for (int g = 0; g < 8; ++g) {
            const int r = g + ((lane & 16) ? 8 : 0);
            lds_nbr[r * NSTR + c0 + (lane & 15)] = acc4[q][g];
        }
    }
    __syncthreads();

    // ---------------- Phase 4: Y = relu((X*diag)@w0^T + nbr@w1^T) ----------------
#pragma unroll
    for (int q = 0; q < 4; ++q) acc4[q] = (v8f){};

    for (int k = 0; k < KSTEPS; ++k) {
        const int k0 = k * 32;
        // Issue all 8 weight B-tile loads up front, then the 8 WMMAs.
        v16bf wb1[4], wb0[4];
#pragma unroll
        for (int q = 0; q < 4; ++q) {
            const int d0 = (wv * 4 + q) * 16;
            wb1[q] = load_b_tile(w1bf, CH, d0, k0);
            wb0[q] = load_b_tile(w0bf, CH, d0, k0);
        }
        v16bf an = a_tile_from_lds_f(lds_nbr, NSTR, k0);             // nbr tile
        v16bf ax = load_a_tile_scaled(Xb, CH, i0, k0, lds_diag);     // X*diag tile
#pragma unroll
        for (int q = 0; q < 4; ++q) acc4[q] = wmma_bf16(an, wb1[q], acc4[q]);
#pragma unroll
        for (int q = 0; q < 4; ++q) acc4[q] = wmma_bf16(ax, wb0[q], acc4[q]);
    }

    // Stage Y as [channel][node] in LDS for coalesced (B,C,N) stores.
#pragma unroll
    for (int q = 0; q < 4; ++q) {
        const int d0 = (wv * 4 + q) * 16;
#pragma unroll
        for (int g = 0; g < 8; ++g) {
            const int i = g + ((lane & 16) ? 8 : 0);
            const int d = d0 + (lane & 15);
            lds_y[d * 16 + i] = fmaxf(acc4[q][g], 0.0f);
        }
    }
    __syncthreads();

    float* ob = out + (long)b * CH * NN + i0;
    for (int t = threadIdx.x; t < CH * 16; t += 128) {
        const int d = t >> 4;
        const int i = t & 15;
        ob[(long)d * NN + i] = lds_y[t];
    }
}

// ---------------------------------------------------------------------------
// Workspace layout (bytes):
//   XbfT : B*N*C bf16   @ 0
//   Xbf  : B*N*C bf16   @ 9,437,184
//   Qbf  : B*N*C bf16   @ 18,874,368
//   Kbf  : B*N*C bf16   @ 28,311,552
//   Wbf  : 4*C*C bf16   @ 37,748,736   (phi, psi, w0, w1 contiguous)
// Total: 38,273,024 bytes
// ---------------------------------------------------------------------------
#define OFF_XT 0L
#define OFF_X  9437184L
#define OFF_Q  18874368L
#define OFF_K  28311552L
#define OFF_W  37748736L

extern "C" void kernel_launch(void* const* d_in, const int* in_sizes, int n_in,
                              void* d_out, int out_size, void* d_ws, size_t ws_size,
                              hipStream_t stream) {
    const float* x   = (const float*)d_in[0];
    const float* phi = (const float*)d_in[1];
    const float* psi = (const float*)d_in[2];
    const float* w0  = (const float*)d_in[3];
    const float* w1  = (const float*)d_in[4];
    float* out = (float*)d_out;

    char* ws = (char*)d_ws;
    __bf16* XbfT  = (__bf16*)(ws + OFF_XT);
    __bf16* Xbf   = (__bf16*)(ws + OFF_X);
    __bf16* Qbf   = (__bf16*)(ws + OFF_Q);
    __bf16* Kbf   = (__bf16*)(ws + OFF_K);
    __bf16* Wbf   = (__bf16*)(ws + OFF_W);
    __bf16* phibf = Wbf;
    __bf16* psibf = Wbf + 1 * CH * CH;
    __bf16* w0bf  = Wbf + 2 * CH * CH;
    __bf16* w1bf  = Wbf + 3 * CH * CH;

    {   // pack x into bf16 X (N,C) and X^T (C,N) via LDS tile transpose
        const int blocks = BATCH * (CH / 32) * (NN / 32);
        pack_x_kernel<<<blocks, 256, 0, stream>>>(x, XbfT, Xbf);
    }
    {   // pack the four weight matrices into bf16
        const int total = 4 * CH * CH;
        pack_w_kernel<<<(total + 255) / 256, 256, 0, stream>>>(phi, psi, w0, w1, Wbf);
    }
    // Q/K projection GEMMs
    qk_gemm_kernel<<<BATCH * ITILES, 128, 0, stream>>>(Xbf, phibf, psibf, Qbf, Kbf);
    // Fused attention + graph message passing + output projection
    fused_graph_kernel<<<BATCH * ITILES, 128, 0, stream>>>(Qbf, Kbf, Xbf, XbfT,
                                                           w0bf, w1bf, out);
}
